// DeepseekV4Indexer_71270687310157
// MI455X (gfx1250) — matrix operand order repaired and checked by reading, not compiled, not run on hardware
//
#include <hip/hip_runtime.h>
#include <hip/hip_bf16.h>
#include <stdint.h>

#define SEQ    4096
#define HID    4096
#define NHEAD  64
#define DHEAD  128
#define NOUT   (NHEAD * DHEAD)   // 8192
#define NCHUNK 1024
#define RCH    4
#define NTOPK  512

// -log2(theta)/32  (rope: inv_freq = theta^(-2j/64) = exp2(j * C))
#define C_ROPE_Q (-0.41524101186092034f)   // theta = 10000
#define C_ROPE_K (-0.54024101186092034f)   // theta = 160000

typedef unsigned short u16;
typedef __attribute__((ext_vector_type(16))) __bf16 bf16x16;
typedef __attribute__((ext_vector_type(8)))  float  f32x8;

union AFrag { bf16x16 v; uint4 q[2]; };
union AccU  { f32x8 v; float f[8]; };

// ---- bf16 round-to-nearest-even ----
__device__ __forceinline__ unsigned bfr(float x) {
  unsigned u = __float_as_uint(x);
  return u + 0x7fffu + ((u >> 16) & 1u);   // bf16 bits = result >> 16
}
#if defined(__has_builtin)
#if __has_builtin(__builtin_amdgcn_cvt_pk_bf16_f32)
#define HAVE_CVT_PK_BF16 1
#endif
#endif
__device__ __forceinline__ unsigned pk2(float lo, float hi) {
#ifdef HAVE_CVT_PK_BF16
  return __builtin_bit_cast(unsigned, __builtin_amdgcn_cvt_pk_bf16_f32(lo, hi));
#else
  return (bfr(lo) >> 16) | (bfr(hi) & 0xffff0000u);
#endif
}
__device__ __forceinline__ u16 f2bf(float x) { return (u16)(bfr(x) >> 16); }

// A fragment (16x32 MxK bf16): lane L<16 -> M=L, K={0..7,16..23}; L>=16 -> M=L-16, K={8..15,24..31}
__device__ __forceinline__ bf16x16 ld_frag_a(const u16* lds, int stride, int mb, int kb, int lane) {
  int m  = mb + (lane & 15);
  int hl = lane >> 4;
  const u16* p = lds + m * stride + kb + hl * 8;
  AFrag f;
  f.q[0] = *(const uint4*)(p);
  f.q[1] = *(const uint4*)(p + 16);
  return f.v;
}
// B fragment (32x16 KxN bf16), LDS layout [n][k]:
// lane L<16 -> N=L, K={0..7,16..23}; L>=16 -> N=L-16, K={8..15,24..31}
__device__ __forceinline__ bf16x16 ld_frag_b(const u16* lds, int stride, int nb, int kb, int lane) {
  int n  = nb + (lane & 15);
  int hl = lane >> 4;
  const u16* p = lds + n * stride + kb + hl * 8;
  AFrag f;
  f.q[0] = *(const uint4*)(p);
  f.q[1] = *(const uint4*)(p + 16);
  return f.v;
}
__device__ __forceinline__ f32x8 wmma_bf16(bf16x16 a, bf16x16 b, f32x8 c) {
  return __builtin_amdgcn_wmma_f32_16x16x32_bf16(false, a, false, b, (short)0, c, false, false);
}

// --- gfx1250 async copy: 16B per lane, GVS addressing (SGPR base + VGPR offset) ---
__device__ __forceinline__ void async_cp16(unsigned lds_addr, unsigned long long gbase, int goff) {
  asm volatile("global_load_async_to_lds_b128 %0, %1, %2"
               :: "v"(lds_addr), "v"(goff), "s"(gbase)
               : "memory");
}
__device__ __forceinline__ void wait_async_le3() {
  asm volatile("s_wait_asynccnt 0x3" ::: "memory");
}
__device__ __forceinline__ void wait_async_le4() {
  asm volatile("s_wait_asynccnt 0x4" ::: "memory");
}
__device__ __forceinline__ void wait_async_0() {
  asm volatile("s_wait_asynccnt 0x0" ::: "memory");
}

// One 64x128 K=32 tile of MMAs. All 4 B fragments are distinct live values so the
// scheduler can issue all ds_loads up front and retire them with partial dscnt waits.
__device__ __forceinline__ void tile_mma(const u16* sAc, const u16* sBc, int strideA,
                                         int strideB, int wm, int wn, int kb, int lane,
                                         AccU* acc) {
  bf16x16 a  = ld_frag_a(sAc, strideA, wm, kb, lane);
  bf16x16 b0 = ld_frag_b(sBc, strideB, wn,      kb, lane);
  bf16x16 b1 = ld_frag_b(sBc, strideB, wn + 16, kb, lane);
  bf16x16 b2 = ld_frag_b(sBc, strideB, wn + 32, kb, lane);
  bf16x16 b3 = ld_frag_b(sBc, strideB, wn + 48, kb, lane);
  acc[0].v = wmma_bf16(a, b0, acc[0].v);
  acc[1].v = wmma_bf16(a, b1, acc[1].v);
  acc[2].v = wmma_bf16(a, b2, acc[2].v);
  acc[3].v = wmma_bf16(a, b3, acc[3].v);
}

// ---------------------------------------------------------------------------
// Shared async double-buffered mainloop: C(64x128) += A(64xK) * B^T(128xK),
// A bf16 [m][k] (row stride HID), B bf16 [n][k] (row stride HID), K = HID.
// Per K-tile: 1 A-chunk + 2 B-chunks per thread via async-to-LDS.
// ---------------------------------------------------------------------------
__device__ __forceinline__ void gemm_mainloop_async(const u16* __restrict__ Abf,
                                                    const u16* __restrict__ Bbf,
                                                    u16 (*sA)[64 * 40], u16 (*sB)[128 * 40],
                                                    int tid, int lane, int wm, int wn,
                                                    AccU* acc) {
  const int ma = tid >> 2, ka = (tid & 3) * 8;   // A: 64 rows x 4 chunks of 8 bf16
  const unsigned long long Ag = (unsigned long long)(const void*)Abf;
  const unsigned long long Bg = (unsigned long long)(const void*)Bbf;
  auto issue = [&](int buf, int k0) {
    async_cp16((unsigned)(uintptr_t)&sA[buf][ma * 40 + ka], Ag, (ma * HID + k0 + ka) * 2);
    async_cp16((unsigned)(uintptr_t)&sB[buf][ma * 40 + ka], Bg, (ma * HID + k0 + ka) * 2);
    async_cp16((unsigned)(uintptr_t)&sB[buf][(ma + 64) * 40 + ka], Bg,
               ((ma + 64) * HID + k0 + ka) * 2);
  };
  issue(0, 0);
  for (int k0 = 0; k0 < HID; k0 += 32) {
    const int cur = (k0 >> 5) & 1;
    if (k0 + 32 < HID) { issue(cur ^ 1, k0 + 32); wait_async_le3(); }
    else wait_async_0();
    __syncthreads();
    tile_mma(sA[cur], sB[cur], 40, 40, wm, wn, 0, lane, acc);
    __syncthreads();
  }
}

// ---------------------------------------------------------------------------
// Format kernels (one-time, bandwidth bound)
// ---------------------------------------------------------------------------
// f32 -> bf16 straight copy; 4 elements / thread. count must be multiple of 1024.
__global__ __launch_bounds__(256) void k_cvt_bf16(const float* __restrict__ src,
                                                  u16* __restrict__ dst) {
  size_t i4 = (size_t)blockIdx.x * 256 + threadIdx.x;
  float4 v = *(const float4*)(src + i4 * 4);
  uint2 o; o.x = pk2(v.x, v.y); o.y = pk2(v.z, v.w);
  *(uint2*)(dst + i4 * 4) = o;
}
// f32 src[K x ncols] -> bf16 dst[ncols x HID] transposed (K == HID), 32x32 LDS tiles.
__global__ __launch_bounds__(256) void k_trans_bf16(const float* __restrict__ src,
                                                    u16* __restrict__ dst, int ncols) {
  __shared__ __align__(16) u16 tile[32 * 34];
  const int k0 = blockIdx.x * 32, n0 = blockIdx.y * 32;
  const int tid = threadIdx.x;
  const int kk = tid >> 3, q = (tid & 7) * 4;
  float4 v = *(const float4*)&src[(size_t)(k0 + kk) * ncols + n0 + q];
  tile[(q + 0) * 34 + kk] = f2bf(v.x);
  tile[(q + 1) * 34 + kk] = f2bf(v.y);
  tile[(q + 2) * 34 + kk] = f2bf(v.z);
  tile[(q + 3) * 34 + kk] = f2bf(v.w);
  __syncthreads();
  const int nn = tid >> 3;
  uint2 o;
  o.x = (unsigned)tile[nn * 34 + q]     | ((unsigned)tile[nn * 34 + q + 1] << 16);
  o.y = (unsigned)tile[nn * 34 + q + 2] | ((unsigned)tile[nn * 34 + q + 3] << 16);
  *(uint2*)&dst[(size_t)(n0 + nn) * HID + k0 + q] = o;
}

// ---------------------------------------------------------------------------
// Kernel 1a: kvgate[s,0:128] = hidden @ wkv ; kvgate[s,128:256] = hidden @ wgate
// (bf16 operands, fully async-staged)
// ---------------------------------------------------------------------------
__global__ __launch_bounds__(256) void k_kvgate_gemm(const u16* __restrict__ hbf,
                                                     const u16* __restrict__ wkvT,
                                                     const u16* __restrict__ wgateT,
                                                     float* __restrict__ kvgate) {
  __shared__ __align__(16) u16 sA[2][64 * 40];
  __shared__ __align__(16) u16 sB[2][128 * 40];
  const int tid = threadIdx.x;
  const int which = blockIdx.x;          // 0 -> wkv, 1 -> wgate
  const int s0 = blockIdx.y * 64;
  const u16* __restrict__ Bbf = which ? wgateT : wkvT;
  const int wid = tid >> 5, lane = tid & 31;
  const int wm = (wid & 3) * 16, wn = (wid >> 2) * 64;
  const f32x8 z8 = {0.f, 0.f, 0.f, 0.f, 0.f, 0.f, 0.f, 0.f};
  AccU acc[4];
#pragma unroll
  for (int t = 0; t < 4; ++t) acc[t].v = z8;

  gemm_mainloop_async(hbf + (size_t)s0 * HID, Bbf, sA, sB, tid, lane, wm, wn, acc);

  const int hl = lane >> 4, nl = lane & 15;
#pragma unroll
  for (int t = 0; t < 4; ++t)
#pragma unroll
    for (int v = 0; v < 8; ++v) {
      int M = wm + v + 8 * hl;
      int N = wn + t * 16 + nl;
      kvgate[(size_t)(s0 + M) * 256 + which * 128 + N] = acc[t].f[v];
    }
}

// ---------------------------------------------------------------------------
// Kernel 1b: softmax-pool over R=4, rmsnorm, partial rope (theta_K) -> k bf16
// ---------------------------------------------------------------------------
__global__ __launch_bounds__(128) void k_poolk(const float* __restrict__ kvgate,
                                               const float* __restrict__ ape,
                                               const float* __restrict__ kvnw,
                                               u16* __restrict__ kbf) {
  __shared__ float red[128];
  __shared__ float stage[128];
  const int n = blockIdx.x, d = threadIdx.x;
  float kvv[RCH], gv[RCH];
#pragma unroll
  for (int r = 0; r < RCH; ++r) {
    int s = n * RCH + r;
    kvv[r] = kvgate[(size_t)s * 256 + d];
    gv[r]  = kvgate[(size_t)s * 256 + 128 + d] + ape[r * DHEAD + d];
  }
  float mx = gv[0];
#pragma unroll
  for (int r = 1; r < RCH; ++r) mx = fmaxf(mx, gv[r]);
  float se = 0.f, pooled = 0.f;
#pragma unroll
  for (int r = 0; r < RCH; ++r) {
    float e = __expf(gv[r] - mx);
    se += e;
    pooled = fmaf(kvv[r], e, pooled);
  }
  pooled /= se;
  red[d] = pooled * pooled;
  __syncthreads();
  for (int off = 64; off > 0; off >>= 1) {
    if (d < off) red[d] += red[d + off];
    __syncthreads();
  }
  float var = red[0] * (1.0f / 128.f);
  float xn = pooled * rsqrtf(var + 1e-6f) * kvnw[d];
  stage[d] = xn;
  __syncthreads();
  float val = xn;
  if (d >= 64) {
    int rr = d - 64, jj = rr >> 1;
    float inv = exp2f((float)jj * C_ROPE_K);
    float ang = (float)(n * RCH) * inv;
    float c, sn;
    __sincosf(ang, &c, &sn);
    float other = stage[d ^ 1];
    val = (rr & 1) ? fmaf(xn, c, other * sn) : fmaf(xn, c, -other * sn);
  }
  kbf[n * DHEAD + d] = f2bf(val);
}

// ---------------------------------------------------------------------------
// Kernel 2: q = q_residual @ wq (bf16 WMMA, async-staged), fused rope, q -> bf16
// ---------------------------------------------------------------------------
__global__ __launch_bounds__(256) void k_qgemm_rope(const u16* __restrict__ qrb,
                                                    const u16* __restrict__ wqT,
                                                    u16* __restrict__ qbf) {
  __shared__ __align__(16) u16 sA[2][64 * 40];
  __shared__ __align__(16) u16 sB[2][128 * 40];
  __shared__ __align__(16) float sC[64 * 132];
  const int tid = threadIdx.x;
  const int n0 = blockIdx.x * 128;       // head = blockIdx.x
  const int s0 = blockIdx.y * 64;
  const int wid = tid >> 5, lane = tid & 31;
  const int wm = (wid & 3) * 16, wn = (wid >> 2) * 64;
  const f32x8 z8 = {0.f, 0.f, 0.f, 0.f, 0.f, 0.f, 0.f, 0.f};
  AccU acc[4];
#pragma unroll
  for (int t = 0; t < 4; ++t) acc[t].v = z8;

  gemm_mainloop_async(qrb + (size_t)s0 * HID, wqT + (size_t)n0 * HID, sA, sB,
                      tid, lane, wm, wn, acc);

  const int hl = lane >> 4, nl = lane & 15;
#pragma unroll
  for (int t = 0; t < 4; ++t)
#pragma unroll
    for (int v = 0; v < 8; ++v) {
      int M = wm + v + 8 * hl;
      int N = wn + t * 16 + nl;
      sC[M * 132 + N] = acc[t].f[v];
    }
  __syncthreads();
  // rope pairs (even,odd) handled per-thread -> packed b32 stores
  for (int p = tid; p < 64 * 64; p += 256) {
    int m = p >> 6, dp = (p & 63) * 2;
    int s = s0 + m;
    float v0 = sC[m * 132 + dp], v1 = sC[m * 132 + dp + 1];
    if (dp >= 64) {
      int jj = (dp - 64) >> 1;
      float inv = exp2f((float)jj * C_ROPE_Q);
      float ang = (float)s * inv, c, sn;
      __sincosf(ang, &c, &sn);
      float e = v0, o = v1;
      v0 = fmaf(e, c, -o * sn);
      v1 = fmaf(o, c,  e * sn);
    }
    *(unsigned*)&qbf[(size_t)s * NOUT + n0 + dp] = pk2(v0, v1);
  }
}

// ---------------------------------------------------------------------------
// Kernel 2b: w = q_residual @ w_weights * NH^-0.5 (LDS-tiled VALU dot)
// ---------------------------------------------------------------------------
__global__ __launch_bounds__(256) void k_wproj(const float* __restrict__ qres,
                                               const float* __restrict__ ww,
                                               float* __restrict__ wout) {
  __shared__ __align__(16) float rows[4 * HID];     // 64 KB
  __shared__ __align__(16) float wtile[64 * 64];    // 16 KB
  const int tid = threadIdx.x;
  const int s0 = blockIdx.x * 4;
#pragma unroll
  for (int j = 0; j < 16; ++j) {
    int i4 = tid + j * 256;
    *(float4*)&rows[i4 * 4] = *(const float4*)&qres[(size_t)s0 * HID + i4 * 4];
  }
  const int r = tid >> 6, h = tid & 63;
  float acc = 0.f;
  for (int kc = 0; kc < HID; kc += 64) {
    __syncthreads();
#pragma unroll
    for (int j = 0; j < 4; ++j) {
      int i4 = tid + j * 256;
      *(float4*)&wtile[i4 * 4] = *(const float4*)&ww[(size_t)kc * NHEAD + i4 * 4];
    }
    __syncthreads();
#pragma unroll
    for (int kk = 0; kk < 64; ++kk)
      acc = fmaf(rows[r * HID + kc + kk], wtile[kk * 64 + h], acc);
  }
  wout[(size_t)(s0 + r) * NHEAD + h] = acc * 0.125f;   // NH^-0.5 = 1/8
}

// ---------------------------------------------------------------------------
// Kernel 3: scores[s,n] = sum_h w[s,h]*relu(q[s,h,:]·k[n,:])*DH^-0.5, causal mask.
// k staged once via async-to-LDS; q double-buffered async across the head loop.
// ---------------------------------------------------------------------------
__global__ __launch_bounds__(256) void k_scores(const u16* __restrict__ qbf,
                                                const u16* __restrict__ kbf,
                                                const float* __restrict__ wproj,
                                                float* __restrict__ scores) {
  __shared__ __align__(16) u16 sK[128 * 136];        // 34 KB
  __shared__ __align__(16) u16 sQ[2][64 * 136];      // 2 x 17 KB
  __shared__ __align__(16) float sW[64 * 64];        // 16 KB
  const int tid = threadIdx.x;
  const int n0 = blockIdx.x * 128;
  const int s0 = blockIdx.y * 64;
  const int wid = tid >> 5, lane = tid & 31;
  const int wm = (wid & 3) * 16, wn = (wid >> 2) * 64;
  const int hl = lane >> 4, nl = lane & 15;

  const unsigned sK_base  = (unsigned)(uintptr_t)(&sK[0]);
  const unsigned sQ0_base = (unsigned)(uintptr_t)(&sQ[0][0]);
  const unsigned sQ1_base = (unsigned)(uintptr_t)(&sQ[1][0]);

  // stage k tile (128 x 128 bf16) via async copies: 8 rounds x 256 lanes x 16B
  {
    unsigned long long gb = (unsigned long long)(const void*)(kbf + (size_t)n0 * DHEAD);
#pragma unroll
    for (int rnd = 0; rnd < 8; ++rnd) {
      int i = tid + rnd * 256;
      int m = i >> 4, c = i & 15;
      async_cp16(sK_base + (unsigned)(m * 136 + c * 8) * 2, gb, (m * DHEAD + c * 8) * 2);
    }
  }
  // stage w block (64 x 64 f32) with vector loads
#pragma unroll
  for (int j = 0; j < 4; ++j) {
    int i4 = tid + j * 256;
    *(float4*)&sW[i4 * 4] = *(const float4*)&wproj[(size_t)s0 * NHEAD + i4 * 4];
  }
  // issue q tile for h=0 into buffer 0
  {
    unsigned long long gb = (unsigned long long)(const void*)(qbf + (size_t)s0 * NOUT);
#pragma unroll
    for (int rnd = 0; rnd < 4; ++rnd) {
      int i = tid + rnd * 256;
      int m = i >> 4, c = i & 15;
      async_cp16(sQ0_base + (unsigned)(m * 136 + c * 8) * 2, gb, (m * NOUT + c * 8) * 2);
    }
  }

  const f32x8 z8 = {0.f, 0.f, 0.f, 0.f, 0.f, 0.f, 0.f, 0.f};
  AccU acc[4];
#pragma unroll
  for (int t = 0; t < 4; ++t) acc[t].v = z8;

  for (int h = 0; h < NHEAD; ++h) {
    const u16* cur = sQ[h & 1];
    if (h + 1 < NHEAD) {
      // prefetch next head's q tile into the other buffer (async, overlapped)
      unsigned nb = ((h + 1) & 1) ? sQ1_base : sQ0_base;
      unsigned long long gb =
          (unsigned long long)(const void*)(qbf + (size_t)s0 * NOUT + (h + 1) * DHEAD);
#pragma unroll
      for (int rnd = 0; rnd < 4; ++rnd) {
        int i = tid + rnd * 256;
        int m = i >> 4, c = i & 15;
        async_cp16(nb + (unsigned)(m * 136 + c * 8) * 2, gb, (m * NOUT + c * 8) * 2);
      }
      wait_async_le4();   // oldest copies (tile h + initial sK) complete; newest 4 in flight
    } else {
      wait_async_0();
    }
    __syncthreads();
    AccU th[4];
#pragma unroll
    for (int t = 0; t < 4; ++t) th[t].v = z8;
#pragma unroll
    for (int ks = 0; ks < 4; ++ks)
      tile_mma(cur, sK, 136, 136, wm, wn, ks * 32, lane, th);
    // per-row w values shared across the 4 N-tiles: load once per head
    float wv[8];
#pragma unroll
    for (int v = 0; v < 8; ++v) wv[v] = sW[(wm + v + 8 * hl) * 64 + h];
#pragma unroll
    for (int t = 0; t < 4; ++t)
#pragma unroll
      for (int v = 0; v < 8; ++v) {
        float sv = fmaxf(th[t].f[v], 0.f) * 0.08838834764831845f;  // DH^-0.5
        acc[t].f[v] = fmaf(sv, wv[v], acc[t].f[v]);
      }
    __syncthreads();  // readers of this buffer done before it is overwritten (h+2)
  }
#pragma unroll
  for (int t = 0; t < 4; ++t)
#pragma unroll
    for (int v = 0; v < 8; ++v) {
      int M = wm + v + 8 * hl, N = wn + t * 16 + nl;
      int s = s0 + M, n = n0 + N;
      float val = acc[t].f[v];
      if (n * RCH + (RCH - 1) > s) val = -1.0e9f;   // causal chunk mask
      scores[(size_t)s * NCHUNK + n] = val;
    }
}

// ---------------------------------------------------------------------------
// Kernel 4: per-row top-512 of 1024 via in-LDS bitonic sort on 64-bit keys.
// ---------------------------------------------------------------------------
__global__ __launch_bounds__(256) void k_topk(const float* __restrict__ scores,
                                              int* __restrict__ topk) {
  __shared__ __align__(16) unsigned long long keys[NCHUNK];
  const int s = blockIdx.x, tid = threadIdx.x;
  for (int i = tid; i < NCHUNK; i += 256) {
    unsigned u = __float_as_uint(scores[(size_t)s * NCHUNK + i]);
    u = (u & 0x80000000u) ? ~u : (u | 0x80000000u);  // order-preserving float->uint
    u = ~u;                                          // invert => ascending sort = descending score
    keys[i] = ((unsigned long long)u << 32) | (unsigned)i;
  }
  __syncthreads();
  for (unsigned k = 2; k <= NCHUNK; k <<= 1) {
    for (unsigned j = k >> 1; j > 0; j >>= 1) {
      for (unsigned t = tid; t < NCHUNK / 2; t += 256) {
        unsigned i = 2 * j * (t / j) + (t % j);
        unsigned ixj = i + j;
        unsigned long long a = keys[i], b = keys[ixj];
        bool up = ((i & k) == 0);
        if ((a > b) == up) { keys[i] = b; keys[ixj] = a; }
      }
      __syncthreads();
    }
  }
  for (int i = tid; i < NTOPK; i += 256)
    topk[(size_t)s * NTOPK + i] = (int)(unsigned)(keys[i] & 0xffffffffu);
}

// ---------------------------------------------------------------------------
extern "C" void kernel_launch(void* const* d_in, const int* in_sizes, int n_in,
                              void* d_out, int out_size, void* d_ws, size_t ws_size,
                              hipStream_t stream) {
  (void)in_sizes; (void)n_in; (void)out_size; (void)ws_size;
  const float* hidden = (const float*)d_in[0];
  const float* qres   = (const float*)d_in[1];
  const float* wq     = (const float*)d_in[2];
  const float* wwts   = (const float*)d_in[3];
  const float* wkv    = (const float*)d_in[4];
  const float* wgate  = (const float*)d_in[5];
  const float* ape    = (const float*)d_in[6];
  const float* kvnw   = (const float*)d_in[7];

  char* ws = (char*)d_ws;
  u16*   abf    = (u16*)ws;    ws += (size_t)SEQ * HID * sizeof(u16);        // 32 MB (hidden, then qres)
  u16*   wqT    = (u16*)ws;    ws += (size_t)NOUT * HID * sizeof(u16);       // 64 MB
  u16*   wkvT   = (u16*)ws;    ws += (size_t)DHEAD * HID * sizeof(u16);      // 1 MB
  u16*   wgateT = (u16*)ws;    ws += (size_t)DHEAD * HID * sizeof(u16);      // 1 MB
  float* kvgate = (float*)ws;  ws += (size_t)SEQ * 256 * sizeof(float);      // 4 MB
  u16*   kbf    = (u16*)ws;    ws += (size_t)NCHUNK * DHEAD * sizeof(u16);   // 256 KB
  u16*   qbf    = (u16*)ws;    ws += (size_t)SEQ * NOUT * sizeof(u16);       // 64 MB
  float* wproj  = (float*)ws;  ws += (size_t)SEQ * NHEAD * sizeof(float);    // 1 MB

  float* scores = (float*)d_out;                         // SEQ x NCHUNK f32
  int*   topk   = (int*)(scores + (size_t)SEQ * NCHUNK); // SEQ x NTOPK i32

  // phase 0: one-time format conversion / transposition (bandwidth bound)
  k_cvt_bf16<<<dim3((SEQ * HID) / 1024), 256, 0, stream>>>(hidden, abf);
  k_trans_bf16<<<dim3(HID / 32, DHEAD / 32), 256, 0, stream>>>(wkv, wkvT, DHEAD);
  k_trans_bf16<<<dim3(HID / 32, DHEAD / 32), 256, 0, stream>>>(wgate, wgateT, DHEAD);
  // phase 1: kv/gate GEMM + pooling -> k
  k_kvgate_gemm<<<dim3(2, SEQ / 64), 256, 0, stream>>>(abf, wkvT, wgateT, kvgate);
  k_poolk<<<dim3(NCHUNK), 128, 0, stream>>>(kvgate, ape, kvnw, kbf);
  // phase 2: q GEMM (abf reused for q_residual bf16)
  k_cvt_bf16<<<dim3((SEQ * HID) / 1024), 256, 0, stream>>>(qres, abf);
  k_trans_bf16<<<dim3(HID / 32, NOUT / 32), 256, 0, stream>>>(wq, wqT, NOUT);
  k_qgemm_rope<<<dim3(NOUT / 128, SEQ / 64), 256, 0, stream>>>(abf, wqT, qbf);
  k_wproj<<<dim3(SEQ / 4), 256, 0, stream>>>(qres, wwts, wproj);
  // phase 3: scores + top-k
  k_scores<<<dim3(NCHUNK / 128, SEQ / 64), 256, 0, stream>>>(qbf, kbf, wproj, scores);
  k_topk<<<dim3(SEQ), 256, 0, stream>>>(scores, topk);
}
